// Fusion_topdown_67465346285930
// MI455X (gfx1250) — compile-verified, matrix operand order checked by reading, not run on hardware
//
#include <hip/hip_runtime.h>
#include <hip/hip_bf16.h>

typedef _Float16 v16h __attribute__((ext_vector_type(16)));
typedef _Float16 v8h  __attribute__((ext_vector_type(8)));
typedef float    v8f  __attribute__((ext_vector_type(8)));
typedef unsigned int v4u __attribute__((ext_vector_type(4)));
typedef int      v8i  __attribute__((ext_vector_type(8)));
typedef int      v4i  __attribute__((ext_vector_type(4)));

#define B_  8
#define C_  128
#define H_  256
#define W_  256
#define D_  64
#define V_  64
#define QL  192

// LDS layout (strides in halves; all multiples of 8 -> 16B-aligned b128 loads)
#define XT_S  264   // X^T  [c=128][h=256]
#define NT_S  136   // ntok [h=256][c=128]
#define K_S   72    // K    [h=256][d=64]
#define Q_S   72    // Q    [q=192][d=64]
#define W_S   136   // WqF/WkF [d=64][c=128]
#define ATT_S 264   // att  [16][h=256] per wave

#define OFF_XT    0         // 128*264*2 = 67584
#define OFF_NT    67584     // max(256*136*2, 8*16*264*2) = 69632 (ntok / attBuf union)
#define OFF_K     137216    // 256*72*2 = 36864
#define OFF_Q     174080    // 192*72*2 = 27648
#define OFF_WQ    201728    // 64*136*2 = 17408
#define OFF_WK    219136    // 17408
#define OFF_WE    236544    // 128*4*4 = 2048
#define OFF_BQ    238592    // 64*4
#define OFF_BK    238848    // 64*4
#define OFF_STAGE 239104    // 2 x 32KB f32 TDM staging [c'=32][h=256]
#define STAGE_BYTES 32768
#define LDS_TOTAL (OFF_STAGE + 2 * STAGE_BYTES)   // 304640 <= 320KB

// ---- Tensor Data Mover: DMA one 32(c) x 256(h) x 1(w) strided slab of x to LDS.
// D# per ISA 8.3-8.6: tile_dim0=1, tile_dim1=256 (stride 256), tile_dim2=32
// (stride 65536), data_size=4B -> LDS receives packed [c'][h] f32 rows.
__device__ __forceinline__ void tdm_load_chunk(const float* gsrc, unsigned lds_off) {
    const unsigned long long ga = (unsigned long long)(uintptr_t)gsrc;
    v4u g0;
    g0[0] = 1u;                                   // count=1, user descriptor
    g0[1] = lds_off;                              // lds_addr (bytes)
    g0[2] = (unsigned)ga;                         // global_addr[31:0]
    g0[3] = (unsigned)((ga >> 32) & 0x01FFFFFFu)  // global_addr[56:32]
          | 0x80000000u;                          // type=2 ("image")
    v8i g1;
    g1[0] = 0x00020000;                           // data_size=2 (4B); mask/pad=0
    g1[1] = 0x00010000;                           // tensor_dim0=1      (bits 79:48)
    g1[2] = 0x01000000;                           // tensor_dim1=256    (bits 111:80)
    g1[3] = 0x00010000;                           // tile_dim0=1        (bits 127:112)
    g1[4] = 0x00200100;                           // tile_dim1=256, tile_dim2=32
    g1[5] = 256;                                  // tensor_dim0_stride (bits 207:160)
    g1[6] = 0;                                    // dim0_stride hi / dim1_stride lo
    g1[7] = 1;                                    // tensor_dim1_stride=65536 (47:16)
    v4i g2;
    g2[0] = 32;                                   // tensor_dim2
    g2[1] = 0; g2[2] = 0; g2[3] = 0;
    v4i g3 = {0, 0, 0, 0};
#if defined(__clang_major__) && (__clang_major__ >= 23)
    v8i g4 = {0, 0, 0, 0, 0, 0, 0, 0};
    __builtin_amdgcn_tensor_load_to_lds(g0, g1, g2, g3, g4, 0);
#else
    __builtin_amdgcn_tensor_load_to_lds(g0, g1, g2, g3, 0);
#endif
}

// Load a 16-half WMMA fragment per ISA 7.12.2 16-bit layout:
// halves 0..7 = K kb..kb+7, halves 8..15 = K kb+16..kb+23 (kb = 0 or 8 by half-wave)
__device__ __forceinline__ v16h frag16(const _Float16* p) {
    v8h lo = *(const v8h*)(p);
    v8h hi = *(const v8h*)(p + 16);
    v16h f;
#pragma unroll
    for (int i = 0; i < 8; ++i) { f[i] = lo[i]; f[i + 8] = hi[i]; }
    return f;
}

__device__ __forceinline__ v8f wmma16(v16h a, v16h b, v8f c) {
    return __builtin_amdgcn_wmma_f32_16x16x32_f16(false, a, false, b, (short)0, c,
                                                  false, false);
}

__global__ __launch_bounds__(256, 1)
void fused_attn_kernel(const float* __restrict__ x,  const float* __restrict__ pe,
                       const float* __restrict__ wemb,
                       const float* __restrict__ gq, const float* __restrict__ bq,
                       const float* __restrict__ wq,
                       const float* __restrict__ gk, const float* __restrict__ bk,
                       const float* __restrict__ wk,
                       float* __restrict__ out) {
    extern __shared__ char smem[];
    _Float16* XT    = (_Float16*)(smem + OFF_XT);
    _Float16* ntok  = (_Float16*)(smem + OFF_NT);
    _Float16* Klds  = (_Float16*)(smem + OFF_K);
    _Float16* Qlds  = (_Float16*)(smem + OFF_Q);
    _Float16* WqF   = (_Float16*)(smem + OFF_WQ);
    _Float16* WkF   = (_Float16*)(smem + OFF_WK);
    float*    weL   = (float*)   (smem + OFF_WE);   // [128][4]
    float*    biasq = (float*)   (smem + OFF_BQ);
    float*    biask = (float*)   (smem + OFF_BK);
    float*    stage = (float*)   (smem + OFF_STAGE);

    const int tid  = threadIdx.x;
    const int lane = tid & 31;
    const int wv   = tid >> 5;
    const int l15  = lane & 15;
    const int hh   = lane >> 4;        // half-wave: 0 or 1
    const int kb   = hh << 3;          // K-base for fragments

    const int b = blockIdx.x >> 8;
    const int w = blockIdx.x & 255;

    const size_t xbase = (size_t)b * C_ * H_ * W_ + (size_t)w;
    const size_t pbase = (size_t)b * 3  * H_ * W_ + (size_t)w;
    const size_t zbase = (size_t)b * C_ * QL * W_ + (size_t)w;

    // kick off TDM chunk 0 (c = 0..31) before the weight stage: overlap DMA+ALU
    if (tid < 32) {
        tdm_load_chunk(x + xbase, (unsigned)(uintptr_t)(smem + OFF_STAGE));
    }

    // ---------- Stage A1: weights into LDS (fold LN gamma into W, beta into bias)
    {
        if (tid < C_) {
            weL[tid * 4 + 0] = wemb[tid * 3 + 0];
            weL[tid * 4 + 1] = wemb[tid * 3 + 1];
            weL[tid * 4 + 2] = wemb[tid * 3 + 2];
        }
        const int d  = tid & 63;
        const int sg = tid >> 6;       // 0..3 -> 32-col chunk
#pragma unroll 8
        for (int c0 = 0; c0 < 32; ++c0) {
            const int c = (sg << 5) + c0;
            WqF[d * W_S + c] = (_Float16)(wq[d * C_ + c] * gq[c]);
            WkF[d * W_S + c] = (_Float16)(wk[d * C_ + c] * gk[c]);
        }
        if (tid < 64) {
            float s = 0.f;
            for (int c = 0; c < C_; ++c) s += bq[c] * wq[tid * C_ + c];
            biasq[tid] = s;
        } else if (tid < 128) {
            const int d2 = tid - 64;
            float s = 0.f;
            for (int c = 0; c < C_; ++c) s += bk[c] * wk[d2 * C_ + c];
            biask[d2] = s;
        }
    }

    // ---------- Stage A2: TDM-staged x slice -> tok / XT / ntok + LN stats
    {
        const int h = tid;  // 256 threads, one h row each
        const float p0 = pe[pbase + (size_t)0 * H_ * W_ + (size_t)h * W_];
        const float p1 = pe[pbase + (size_t)1 * H_ * W_ + (size_t)h * W_];
        const float p2 = pe[pbase + (size_t)2 * H_ * W_ + (size_t)h * W_];
        float sum = 0.f, ss = 0.f;
        for (int ch = 0; ch < 4; ++ch) {            // 4 chunks of 32 c-rows
            if (tid < 32) {
                if (ch + 1 < 4) {                    // prefetch next chunk (other buf)
                    tdm_load_chunk(x + xbase + (size_t)(ch + 1) * 32 * H_ * W_,
                                   (unsigned)(uintptr_t)(smem + OFF_STAGE
                                        + ((ch + 1) & 1) * STAGE_BYTES));
                    // in-order completion: <=1 outstanding => chunk `ch` landed
                    __builtin_amdgcn_s_wait_tensorcnt(1);
                } else {
                    __builtin_amdgcn_s_wait_tensorcnt(0);
                }
            }
            __syncthreads();                         // publish chunk `ch` to all waves
            const float* sb = stage + (ch & 1) * (STAGE_BYTES / 4);
#pragma unroll 8
            for (int c0 = 0; c0 < 32; ++c0) {
                const int   c    = (ch << 5) + c0;
                const float xv   = sb[c0 * H_ + h];  // bank-conflict-free (addr%64==h%64)
                const float tokv = xv + weL[c * 4 + 0] * p0 + weL[c * 4 + 1] * p1
                                      + weL[c * 4 + 2] * p2;
                XT[c * XT_S + h]   = (_Float16)xv;   // raw x for att@X
                ntok[h * NT_S + c] = (_Float16)tokv; // tok for LN+proj
                sum += tokv; ss += tokv * tokv;
            }
            __syncthreads();                         // before buffer (ch&1) is reused
        }
        const float mu  = sum * (1.f / 128.f);
        const float var = ss * (1.f / 128.f) - mu * mu;
        const float rs  = rsqrtf(var + 1e-5f);
        for (int c = 0; c < C_; ++c) {
            const float v = (float)ntok[h * NT_S + c];
            ntok[h * NT_S + c] = (_Float16)((v - mu) * rs);
        }
    }
    __syncthreads();

    // ---------- Stage B: projections  K = ntok @ WkFᵀ (+biask), Q likewise
    // 28 row-tiles: 0..15 -> K (h), 16..27 -> Q (h = 64 + 16*(t-16))
    for (int tile = wv; tile < 28; tile += 8) {
        const bool isQ  = tile >= 16;
        const int  mrow = isQ ? (64 + ((tile - 16) << 4) + l15) : ((tile << 4) + l15);
        const _Float16* arow = ntok + mrow * NT_S;
        const _Float16* wmat = isQ ? WqF : WkF;
        const float*    bias = isQ ? biasq : biask;
        v8f acc[4];
#pragma unroll
        for (int nt = 0; nt < 4; ++nt) acc[nt] = (v8f){0,0,0,0,0,0,0,0};
#pragma unroll
        for (int ks = 0; ks < 4; ++ks) {
            const v16h a = frag16(arow + (ks << 5) + kb);
#pragma unroll
            for (int nt = 0; nt < 4; ++nt) {
                const v16h bf = frag16(wmat + ((nt << 4) + l15) * W_S + (ks << 5) + kb);
                acc[nt] = wmma16(a, bf, acc[nt]);
            }
        }
#pragma unroll
        for (int nt = 0; nt < 4; ++nt) {
            const float bv = bias[(nt << 4) + l15];
#pragma unroll
            for (int v = 0; v < 8; ++v) {
                const int   m   = v + (hh << 3);
                const float val = acc[nt][v] + bv;
                if (isQ)
                    Qlds[(((tile - 16) << 4) + m) * Q_S + (nt << 4) + l15] = (_Float16)val;
                else
                    Klds[((tile << 4) + m) * K_S + (nt << 4) + l15] = (_Float16)val;
            }
        }
    }
    __syncthreads();   // also: attBuf aliases ntok from here on

    // ---------- Stage C: attention, one 16-row q-tile per wave iteration
    _Float16* ab = (_Float16*)(smem + OFF_NT) + wv * 16 * ATT_S;  // per-wave att tile
    for (int qt = wv; qt < 12; qt += 8) {
        // S = Q Kᵀ   (16 x 256)
        v8f S[16];
#pragma unroll
        for (int ht = 0; ht < 16; ++ht) S[ht] = (v8f){0,0,0,0,0,0,0,0};
        const _Float16* qrow = Qlds + ((qt << 4) + l15) * Q_S;
#pragma unroll
        for (int ks = 0; ks < 2; ++ks) {
            const v16h a = frag16(qrow + (ks << 5) + kb);
#pragma unroll
            for (int ht = 0; ht < 16; ++ht) {
                const v16h bf = frag16(Klds + ((ht << 4) + l15) * K_S + (ks << 5) + kb);
                S[ht] = wmma16(a, bf, S[ht]);
            }
        }
        // banded-causal mask: valid iff h <= q + 64
#pragma unroll
        for (int ht = 0; ht < 16; ++ht) {
            const int h = (ht << 4) + l15;
#pragma unroll
            for (int v = 0; v < 8; ++v) {
                const int q = (qt << 4) + v + (hh << 3);
                if (h > q + V_) S[ht][v] = -1.0e9f;
            }
        }
        // softmax per row (row m = v + 8*hh lives in one 16-lane half)
#pragma unroll
        for (int v = 0; v < 8; ++v) {
            float mx = -3.0e38f;
#pragma unroll
            for (int ht = 0; ht < 16; ++ht) mx = fmaxf(mx, S[ht][v]);
#pragma unroll
            for (int off = 1; off < 16; off <<= 1) mx = fmaxf(mx, __shfl_xor(mx, off));
            float sum = 0.f;
#pragma unroll
            for (int ht = 0; ht < 16; ++ht) {
                const float p = __expf(S[ht][v] - mx);
                S[ht][v] = p; sum += p;
            }
#pragma unroll
            for (int off = 1; off < 16; off <<= 1) sum += __shfl_xor(sum, off);
            const float inv = 1.0f / sum;
            const int   m   = v + (hh << 3);
#pragma unroll
            for (int ht = 0; ht < 16; ++ht)
                ab[m * ATT_S + (ht << 4) + l15] = (_Float16)(S[ht][v] * inv);
        }
        // Z = att @ X   (16 x 128), X^T[c][h] is the B operand [n][k]
        v8f Z[8];
#pragma unroll
        for (int nt = 0; nt < 8; ++nt) Z[nt] = (v8f){0,0,0,0,0,0,0,0};
        const _Float16* arow2 = ab + l15 * ATT_S;
#pragma unroll
        for (int ks = 0; ks < 8; ++ks) {
            const v16h a = frag16(arow2 + (ks << 5) + kb);
#pragma unroll
            for (int nt = 0; nt < 8; ++nt) {
                const v16h bf = frag16(XT + ((nt << 4) + l15) * XT_S + (ks << 5) + kb);
                Z[nt] = wmma16(a, bf, Z[nt]);
            }
        }
        // scatter z[b][c][q][w]
#pragma unroll
        for (int nt = 0; nt < 8; ++nt) {
            const int c = (nt << 4) + l15;
#pragma unroll
            for (int v = 0; v < 8; ++v) {
                const int q = (qt << 4) + v + (hh << 3);
                out[zbase + ((size_t)c * QL + q) * W_] = Z[nt][v];
            }
        }
    }
}

extern "C" void kernel_launch(void* const* d_in, const int* in_sizes, int n_in,
                              void* d_out, int out_size, void* d_ws, size_t ws_size,
                              hipStream_t stream) {
    const float* x    = (const float*)d_in[0];
    const float* pe   = (const float*)d_in[1];
    const float* wemb = (const float*)d_in[2];
    const float* gq   = (const float*)d_in[3];
    const float* bq   = (const float*)d_in[4];
    const float* wq   = (const float*)d_in[5];
    const float* gk   = (const float*)d_in[6];
    const float* bk   = (const float*)d_in[7];
    const float* wk   = (const float*)d_in[8];
    float* out = (float*)d_out;

    dim3 grid(B_ * W_);   // 2048 workgroups: one (b,w) sequence each, w-major
    dim3 block(256);      // 8 wave32
    fused_attn_kernel<<<grid, block, LDS_TOTAL, stream>>>(
        x, pe, wemb, gq, bq, wq, gk, bk, wk, out);
}
